// XFMR_12627203851172
// MI455X (gfx1250) — matrix operand order repaired and checked by reading, not compiled, not run on hardware
//
#include <hip/hip_runtime.h>
#include <hip/hip_bf16.h>
#include <stdint.h>

// ---------------------------------------------------------------------------
// MI455X (gfx1250) fused transformer block: bf16 WMMA matrix math,
// TDM async tile loads (tensor_load_to_lds), double-buffered LDS GEMM.
// ---------------------------------------------------------------------------

typedef __bf16 bf16_t;
typedef __attribute__((ext_vector_type(16))) __bf16 v16bf;
typedef __attribute__((ext_vector_type(8)))  float  v8f;

#define WMMA_BF16(a, b, c) \
  __builtin_amdgcn_wmma_f32_16x16x32_bf16(false, (a), false, (b), (short)0, (c), false, false)

// Constants from the reference problem.
#define N_TOK   65536
#define C_DIM   256
#define HID_DIM 1024
#define HEADS   8
#define DH      32
#define BUCK    256
#define NB      (N_TOK / BUCK)
#define SHIFT_R 128

// Tensor Data Mover availability (device pass only; host pass parses fallback).
#if defined(__AMDGCN__) && __has_builtin(__builtin_amdgcn_tensor_load_to_lds)
#define USE_TDM 1
#else
#define USE_TDM 0
#endif

#if USE_TDM
typedef __attribute__((ext_vector_type(4))) unsigned int v4u_t;
typedef __attribute__((ext_vector_type(4))) int v4i_t;
typedef __attribute__((ext_vector_type(8))) int v8i_t;

// Issue a 2D TDM load: tile_k (dim0, elems) x tile_rows (dim1), 2-byte elems,
// row stride = row_stride_elems, into LDS byte offset lds_off.
// D# packing per CDNA5 ISA 8.3 / 8.4 (group0: count/lds/global/type,
// group1: data_size, tensor dims, tile dims, dim0 stride).
__device__ __forceinline__ void tdm_load_2d(unsigned lds_off, const void* gaddr,
                                            unsigned tile_k, unsigned tile_rows,
                                            unsigned row_stride_elems,
                                            unsigned tensor_k) {
  unsigned long long ga = (unsigned long long)(uintptr_t)gaddr;
  v4u_t g0;
  g0[0] = 1u;                                                 // count=1, user D#
  g0[1] = lds_off;                                            // lds_addr (bytes)
  g0[2] = (unsigned)ga;                                       // global_addr[31:0]
  g0[3] = (unsigned)((ga >> 32) & 0x01FFFFFFu) | (2u << 30);  // addr[56:32] | type=2
  v8i_t g1;
  g1[0] = (int)(1u << 16);                                    // data_size = 2 bytes
  g1[1] = (int)((tensor_k & 0xFFFFu) << 16);                  // tensor_dim0[15:0]
  g1[2] = (int)((tensor_k >> 16) & 0xFFFFu);                  // dim0[31:16] | dim1 lo (=0)
  g1[3] = (int)0x0010u | (int)((tile_k & 0xFFFFu) << 16);     // dim1 hi (1M rows) | tile_dim0
  g1[4] = (int)(tile_rows & 0xFFFFu);                         // tile_dim1 | tile_dim2=0
  g1[5] = (int)row_stride_elems;                              // tensor_dim0_stride[31:0]
  g1[6] = 0;                                                  // stride hi | dim1_stride lo
  g1[7] = 0;
  v4i_t gz = {0, 0, 0, 0};
#if __clang_major__ >= 23
  v8i_t gz8 = {0, 0, 0, 0, 0, 0, 0, 0};
  __builtin_amdgcn_tensor_load_to_lds(g0, g1, gz, gz, gz8, 0);
#else
  __builtin_amdgcn_tensor_load_to_lds(g0, g1, gz, gz, 0);
#endif
}
#endif  // USE_TDM

// ---------------------------------------------------------------------------
// Fragment loaders (contiguous k order, not the ISA table order, 16-bit data, wave32).
// A 16x32: lanes 0-15 row M=lane, V0..7=K{0..15};
//          lanes 16-31 K{16..31}.
// ---------------------------------------------------------------------------
__device__ __forceinline__ v16bf load_fragA(const bf16_t* base, int ld) {
  const int lane = threadIdx.x & 31;
  const int r    = lane & 15;
  const int kh   = (lane & 16) ? 16 : 0;
  const bf16_t* p = base + r * ld + kh;
  v16bf f;
  unsigned int* fd = reinterpret_cast<unsigned int*>(&f);
#pragma unroll
  for (int j = 0; j < 8; ++j) {
    fd[j]     = *reinterpret_cast<const unsigned int*>(p + 2 * j);
  }
  return f;
}

// B 32x16 (KxN) from an LDS tile holding B^T as [n][k]:
// lanes 0-15 hold K=0..15 (2/VGPR), lanes 16-31 hold K=16..31.
__device__ __forceinline__ v16bf load_fragB(const bf16_t* baseT, int ld) {
  const int lane = threadIdx.x & 31;
  const int n    = lane & 15;
  const int kb   = (lane & 16) ? 16 : 0;
  const bf16_t* p = baseT + n * ld + kb;
  v16bf f;
  unsigned int* fd = reinterpret_cast<unsigned int*>(&f);
#pragma unroll
  for (int j = 0; j < 8; ++j)
    fd[j] = *reinterpret_cast<const unsigned int*>(p + 2 * j);
  return f;
}

__device__ __forceinline__ float gelu_tanh(float x) {
  const float k0 = 0.7978845608028654f;   // sqrt(2/pi)
  const float k1 = 0.044715f;
  float x3 = x * x * x;
  return 0.5f * x * (1.0f + tanhf(k0 * (x + k1 * x3)));
}

// ---------------------------------------------------------------------------
// fp32 -> bf16 elementwise convert (weights).
// ---------------------------------------------------------------------------
__global__ void cvt_f32_bf16(const float* __restrict__ in, bf16_t* __restrict__ out, int n) {
  int i = blockIdx.x * blockDim.x + threadIdx.x;
  if (i < n) out[i] = (bf16_t)in[i];
}

// ---------------------------------------------------------------------------
// LayerNorm over C=256 with row roll: out[i] = LN(x[(i+roll) mod N]).
// One wave32 per row, 8 elements per lane, shfl_xor reductions.
// ---------------------------------------------------------------------------
__global__ __launch_bounds__(256) void layernorm_bf16(
    const float* __restrict__ x, const float* __restrict__ g,
    const float* __restrict__ b, bf16_t* __restrict__ out, int roll) {
  const int lane = threadIdx.x & 31;
  const int w    = threadIdx.x >> 5;
  const int row  = blockIdx.x * 8 + w;
  int src = row + roll;
  if (src >= N_TOK) src -= N_TOK;
  const float* xr = x + (size_t)src * C_DIM;

  float v[8];
  float s = 0.f, s2 = 0.f;
#pragma unroll
  for (int j = 0; j < 8; ++j) {
    v[j] = xr[lane + 32 * j];
    s  += v[j];
    s2 += v[j] * v[j];
  }
#pragma unroll
  for (int m = 16; m >= 1; m >>= 1) {
    s  += __shfl_xor(s,  m, 32);
    s2 += __shfl_xor(s2, m, 32);
  }
  const float mu  = s * (1.0f / C_DIM);
  const float var = s2 * (1.0f / C_DIM) - mu * mu;
  const float rs  = rsqrtf(var + 1e-5f);
#pragma unroll
  for (int j = 0; j < 8; ++j) {
    int c = lane + 32 * j;
    out[(size_t)row * C_DIM + c] = (bf16_t)((v[j] - mu) * rs * g[c] + b[c]);
  }
}

// ---------------------------------------------------------------------------
// Double-buffered bf16 GEMM, 128x128x32 stages, 8 wave32s, 2x4 accums/wave.
// A-tile arrives via TDM (async, TENSORcnt) while B-tile is staged transposed
// by the VALU path; compute overlaps the next stage's loads.
// ---------------------------------------------------------------------------
template <bool DO_GELU, bool DO_RESID, bool OUT_BF16>
__global__ __launch_bounds__(256) void gemm_bf16(
    const bf16_t* __restrict__ A, const bf16_t* __restrict__ B,
    const float* __restrict__ bias, const float* __restrict__ resid,
    float* __restrict__ outF, bf16_t* __restrict__ outB,
    int M, int Nc, int K, int storeShift) {
  __shared__ alignas(16) bf16_t As[2][128 * 32];   // [m][k] ping-pong
  __shared__ alignas(16) bf16_t Bs[2][128 * 32];   // [n][k] (B^T) ping-pong

  const int tid  = threadIdx.x;
  const int lane = tid & 31;
  const int w    = tid >> 5;
  const int bm   = blockIdx.y * 128;
  const int bn   = blockIdx.x * 128;
  const int wm   = (w >> 1) * 32;   // 4 wave-rows
  const int wn   = (w & 1) * 64;    // 2 wave-cols
  const int nk   = K / 32;

  v8f acc[2][4];
#pragma unroll
  for (int i = 0; i < 2; ++i)
#pragma unroll
    for (int j = 0; j < 4; ++j) acc[i][j] = (v8f){0.f, 0.f, 0.f, 0.f, 0.f, 0.f, 0.f, 0.f};

  const int arow  = tid >> 1, akoff = (tid & 1) * 16;   // A fallback: 16 elems each
  const int bkrow = tid >> 3, bnoff = (tid & 7) * 16;   // B: 16 elems each

  auto load_B = [&](int buf, int k0) {
    const bf16_t* src = B + (size_t)(k0 + bkrow) * Nc + bn + bnoff;
    if (k0 + 64 < K) __builtin_prefetch(src + (size_t)64 * Nc, 0, 3);  // 2 stages ahead
#pragma unroll
    for (int t = 0; t < 16; ++t) Bs[buf][(bnoff + t) * 32 + bkrow] = src[t];
  };

#if USE_TDM
  auto load_A = [&](int buf, int k0) {
    if (w == 0)
      tdm_load_2d((unsigned)(uintptr_t)&As[buf][0], A + (size_t)bm * K + k0,
                  /*tile_k=*/32, /*tile_rows=*/128,
                  /*row_stride=*/(unsigned)K, /*tensor_k=*/(unsigned)K);
  };
  auto wait_A = [&]() {
    if (w == 0) __builtin_amdgcn_s_wait_tensorcnt(0);
  };
#else
  auto load_A = [&](int buf, int k0) {
    const bf16_t* src = A + (size_t)(bm + arow) * K + k0 + akoff;
    if (k0 + 64 < K) __builtin_prefetch(src + 64, 0, 3);
    *reinterpret_cast<uint4*>(&As[buf][arow * 32 + akoff]) =
        *reinterpret_cast<const uint4*>(src);
    *reinterpret_cast<uint4*>(&As[buf][arow * 32 + akoff + 8]) =
        *reinterpret_cast<const uint4*>(src + 8);
  };
  auto wait_A = [&]() {};
#endif

  // Pipeline prologue: stage 0 in flight.
  load_A(0, 0);
  load_B(0, 0);
  wait_A();
  __syncthreads();

  for (int s = 0; s < nk; ++s) {
    const int buf = s & 1;
    if (s + 1 < nk) {               // overlap next stage with compute
      load_A(buf ^ 1, (s + 1) * 32);
      load_B(buf ^ 1, (s + 1) * 32);
    }

    v16bf a0 = load_fragA(&As[buf][(wm + 0) * 32], 32);
    v16bf a1 = load_fragA(&As[buf][(wm + 16) * 32], 32);
#pragma unroll
    for (int j = 0; j < 4; ++j) {
      v16bf bfb = load_fragB(&Bs[buf][(wn + j * 16) * 32], 32);
      acc[0][j] = WMMA_BF16(a0, bfb, acc[0][j]);
      acc[1][j] = WMMA_BF16(a1, bfb, acc[1][j]);
    }

    if (s + 1 < nk) wait_A();       // TDM landed before everyone passes barrier
    __syncthreads();
  }

  // Epilogue: D layout — n = lane&15, m = v + 8*(lane>=16).
  const int nloc = lane & 15;
  const int hi   = (lane & 16) ? 8 : 0;
#pragma unroll
  for (int i = 0; i < 2; ++i)
#pragma unroll
    for (int j = 0; j < 4; ++j) {
      const int n  = bn + wn + j * 16 + nloc;
      const float bv = bias[n];
#pragma unroll
      for (int v = 0; v < 8; ++v) {
        int m = bm + wm + i * 16 + v + hi;
        int orow = m + storeShift;
        if (orow >= M) orow -= M;
        float val = acc[i][j][v] + bv;
        if (DO_GELU) val = gelu_tanh(val);
        if (DO_RESID) val += resid[(size_t)orow * Nc + n];
        if (OUT_BF16) outB[(size_t)orow * Nc + n] = (bf16_t)val;
        else          outF[(size_t)orow * Nc + n] = val;
      }
    }
}

// ---------------------------------------------------------------------------
// Bucketed attention, one workgroup per (head, bucket). q/k/v^T in LDS;
// each wave owns 32 query rows; two-pass softmax (row-max, then exp/sum + PV).
// dynamic LDS: q 16K + k 16K + v^T 16K + per-wave P staging 32K = 80 KB.
// ---------------------------------------------------------------------------
__global__ __launch_bounds__(256) void bucket_attn(
    const bf16_t* __restrict__ qkv, bf16_t* __restrict__ out) {
  extern __shared__ char smem[];
  bf16_t* qs  = reinterpret_cast<bf16_t*>(smem);  // [256][32]
  bf16_t* ks  = qs + 256 * 32;                    // [256][32]
  bf16_t* vts = ks + 256 * 32;                    // [32][256]  (v transposed)
  bf16_t* pb  = vts + 32 * 256;                   // [8][32][64]

  const int head = blockIdx.x;
  const int buck = blockIdx.y;
  const int tid  = threadIdx.x;
  const int lane = tid & 31;
  const int w    = tid >> 5;
  const size_t tok0 = (size_t)buck * BUCK;

  // Stage q/k into LDS; v stored transposed for B-fragment loads.
  {
    const bf16_t* base = qkv + (tok0 + tid) * (3 * C_DIM) + head * DH;
    *reinterpret_cast<uint4*>(&qs[tid * 32 + 0])  = *reinterpret_cast<const uint4*>(base + 0);
    *reinterpret_cast<uint4*>(&qs[tid * 32 + 8])  = *reinterpret_cast<const uint4*>(base + 8);
    *reinterpret_cast<uint4*>(&qs[tid * 32 + 16]) = *reinterpret_cast<const uint4*>(base + 16);
    *reinterpret_cast<uint4*>(&qs[tid * 32 + 24]) = *reinterpret_cast<const uint4*>(base + 24);
    const bf16_t* kbse = base + C_DIM;
    *reinterpret_cast<uint4*>(&ks[tid * 32 + 0])  = *reinterpret_cast<const uint4*>(kbse + 0);
    *reinterpret_cast<uint4*>(&ks[tid * 32 + 8])  = *reinterpret_cast<const uint4*>(kbse + 8);
    *reinterpret_cast<uint4*>(&ks[tid * 32 + 16]) = *reinterpret_cast<const uint4*>(kbse + 16);
    *reinterpret_cast<uint4*>(&ks[tid * 32 + 24]) = *reinterpret_cast<const uint4*>(kbse + 24);
    const bf16_t* vbse = base + 2 * C_DIM;
#pragma unroll
    for (int d = 0; d < DH; ++d) vts[d * 256 + tid] = vbse[d];
  }
  __syncthreads();

  const float scale = 0.17677669529663687f;  // 1/sqrt(32)
  const int qb = w * 32;
  const int nloc = lane & 15;
  const int hi   = (lane & 16) ? 8 : 0;

  v16bf aq0 = load_fragA(qs + (qb + 0) * 32, 32);
  v16bf aq1 = load_fragA(qs + (qb + 16) * 32, 32);
  const v8f z = (v8f){0.f, 0.f, 0.f, 0.f, 0.f, 0.f, 0.f, 0.f};

  // ---- Pass 1: row max over 256 keys ----
  float rmax[2][8];
#pragma unroll
  for (int i = 0; i < 2; ++i)
#pragma unroll
    for (int v = 0; v < 8; ++v) rmax[i][v] = -1e30f;

  for (int c = 0; c < 4; ++c) {
#pragma unroll
    for (int jj = 0; jj < 4; ++jj) {
      v16bf bk = load_fragB(ks + (c * 64 + jj * 16) * 32, 32);
      v8f s0 = WMMA_BF16(aq0, bk, z);
      v8f s1 = WMMA_BF16(aq1, bk, z);
#pragma unroll
      for (int v = 0; v < 8; ++v) {
        rmax[0][v] = fmaxf(rmax[0][v], s0[v] * scale);
        rmax[1][v] = fmaxf(rmax[1][v], s1[v] * scale);
      }
    }
  }
#pragma unroll
  for (int i = 0; i < 2; ++i)
#pragma unroll
    for (int v = 0; v < 8; ++v) {
      float m = rmax[i][v];
      m = fmaxf(m, __shfl_xor(m, 1, 32));
      m = fmaxf(m, __shfl_xor(m, 2, 32));
      m = fmaxf(m, __shfl_xor(m, 4, 32));
      m = fmaxf(m, __shfl_xor(m, 8, 32));
      rmax[i][v] = m;
    }

  // ---- Pass 2: exp, row sums, and P@V accumulation ----
  float rsum[2][8];
#pragma unroll
  for (int i = 0; i < 2; ++i)
#pragma unroll
    for (int v = 0; v < 8; ++v) rsum[i][v] = 0.f;

  v8f o[2][2];
#pragma unroll
  for (int i = 0; i < 2; ++i)
#pragma unroll
    for (int j = 0; j < 2; ++j) o[i][j] = z;

  bf16_t* pw = pb + w * 32 * 64;  // this wave's [32 q][64 key] staging tile

  for (int c = 0; c < 4; ++c) {
#pragma unroll
    for (int jj = 0; jj < 4; ++jj) {
      v16bf bk = load_fragB(ks + (c * 64 + jj * 16) * 32, 32);
      v8f s0 = WMMA_BF16(aq0, bk, z);
      v8f s1 = WMMA_BF16(aq1, bk, z);
#pragma unroll
      for (int v = 0; v < 8; ++v) {
        float p0 = __expf(s0[v] * scale - rmax[0][v]);
        float p1 = __expf(s1[v] * scale - rmax[1][v]);
        rsum[0][v] += p0;
        rsum[1][v] += p1;
        pw[(0 * 16 + v + hi) * 64 + jj * 16 + nloc] = (bf16_t)p0;
        pw[(1 * 16 + v + hi) * 64 + jj * 16 + nloc] = (bf16_t)p1;
      }
    }
    __syncthreads();  // D-layout stores visible before A-layout reloads
#pragma unroll
    for (int kk = 0; kk < 2; ++kk) {
      v16bf pa0 = load_fragA(pw + 0 * 64 + kk * 32, 64);
      v16bf pa1 = load_fragA(pw + 16 * 64 + kk * 32, 64);
#pragma unroll
      for (int jn = 0; jn < 2; ++jn) {
        v16bf bv = load_fragB(vts + (jn * 16) * 256 + c * 64 + kk * 32, 256);
        o[0][jn] = WMMA_BF16(pa0, bv, o[0][jn]);
        o[1][jn] = WMMA_BF16(pa1, bv, o[1][jn]);
      }
    }
    __syncthreads();
  }

#pragma unroll
  for (int i = 0; i < 2; ++i)
#pragma unroll
    for (int v = 0; v < 8; ++v) {
      float s = rsum[i][v];
      s += __shfl_xor(s, 1, 32);
      s += __shfl_xor(s, 2, 32);
      s += __shfl_xor(s, 4, 32);
      s += __shfl_xor(s, 8, 32);
      rsum[i][v] = s;
    }

#pragma unroll
  for (int i = 0; i < 2; ++i)
#pragma unroll
    for (int jn = 0; jn < 2; ++jn)
#pragma unroll
      for (int v = 0; v < 8; ++v) {
        int row = qb + i * 16 + v + hi;
        int col = head * DH + jn * 16 + nloc;
        out[(tok0 + row) * C_DIM + col] = (bf16_t)(o[i][jn][v] / rsum[i][v]);
      }
}

// ---------------------------------------------------------------------------
// Host-side orchestration.
// ---------------------------------------------------------------------------
extern "C" void kernel_launch(void* const* d_in, const int* in_sizes, int n_in,
                              void* d_out, int out_size, void* d_ws, size_t ws_size,
                              hipStream_t stream) {
  (void)in_sizes; (void)n_in; (void)out_size; (void)ws_size;

  const float* feat  = (const float*)d_in[1];
  const float* ln1w  = (const float*)d_in[3];
  const float* ln1b  = (const float*)d_in[4];
  const float* wqkv  = (const float*)d_in[5];
  const float* bqkv  = (const float*)d_in[6];
  const float* wo    = (const float*)d_in[7];
  const float* bo    = (const float*)d_in[8];
  const float* ln2w  = (const float*)d_in[9];
  const float* ln2b  = (const float*)d_in[10];
  const float* w1    = (const float*)d_in[11];
  const float* b1    = (const float*)d_in[12];
  const float* w2    = (const float*)d_in[13];
  const float* b2    = (const float*)d_in[14];
  float* outp = (float*)d_out;

  // Workspace carve-up (256-byte aligned regions).
  char* p = (char*)d_ws;
  auto carve = [&](size_t bytes) -> char* {
    char* r = p;
    p += (bytes + 255) & ~(size_t)255;
    return r;
  };
  bf16_t* xs    = (bf16_t*)carve((size_t)N_TOK * C_DIM * 2);     // LN1 (rolled); reused as attn_out
  bf16_t* qkvb  = (bf16_t*)carve((size_t)N_TOK * 3 * C_DIM * 2);
  float*  mid   = (float*) carve((size_t)N_TOK * C_DIM * 4);
  bf16_t* h2    = (bf16_t*)carve((size_t)N_TOK * C_DIM * 2);
  bf16_t* ffn1  = (bf16_t*)carve((size_t)N_TOK * HID_DIM * 2);
  bf16_t* wqkvB = (bf16_t*)carve((size_t)C_DIM * 3 * C_DIM * 2);
  bf16_t* woB   = (bf16_t*)carve((size_t)C_DIM * C_DIM * 2);
  bf16_t* w1B   = (bf16_t*)carve((size_t)C_DIM * HID_DIM * 2);
  bf16_t* w2B   = (bf16_t*)carve((size_t)HID_DIM * C_DIM * 2);
  bf16_t* attn_out = xs;  // xs dead after QKV GEMM

  // 1) Weights -> bf16.
  cvt_f32_bf16<<<(C_DIM * 3 * C_DIM + 255) / 256, 256, 0, stream>>>(wqkv, wqkvB, C_DIM * 3 * C_DIM);
  cvt_f32_bf16<<<(C_DIM * C_DIM + 255) / 256, 256, 0, stream>>>(wo, woB, C_DIM * C_DIM);
  cvt_f32_bf16<<<(C_DIM * HID_DIM + 255) / 256, 256, 0, stream>>>(w1, w1B, C_DIM * HID_DIM);
  cvt_f32_bf16<<<(HID_DIM * C_DIM + 255) / 256, 256, 0, stream>>>(w2, w2B, HID_DIM * C_DIM);

  // 2) LN1 with roll(-SHIFT) folded into the gather: xs[i] = LN(feat[(i+128) mod N]).
  layernorm_bf16<<<N_TOK / 8, 256, 0, stream>>>(feat, ln1w, ln1b, xs, SHIFT_R);

  // 3) QKV projection: [N,256] x [256,768] -> bf16.
  gemm_bf16<false, false, true><<<dim3(768 / 128, N_TOK / 128), 256, 0, stream>>>(
      xs, wqkvB, bqkv, nullptr, nullptr, qkvb, N_TOK, 3 * C_DIM, C_DIM, 0);

  // 4) Bucketed attention, one block per (head, bucket). 80 KB dynamic LDS.
  const size_t attn_lds = (size_t)(256 * 32 + 256 * 32 + 32 * 256 + 8 * 32 * 64) * 2;
  bucket_attn<<<dim3(HEADS, NB), 256, attn_lds, stream>>>(qkvb, attn_out);

  // 5) Out projection + roll(+SHIFT) scatter + residual: mid = roll(attn@Wo+bo, +128) + feat.
  gemm_bf16<false, true, false><<<dim3(C_DIM / 128, N_TOK / 128), 256, 0, stream>>>(
      attn_out, woB, bo, feat, mid, nullptr, N_TOK, C_DIM, C_DIM, SHIFT_R);

  // 6) LN2.
  layernorm_bf16<<<N_TOK / 8, 256, 0, stream>>>(mid, ln2w, ln2b, h2, 0);

  // 7) FFN1 + gelu (bf16 out).
  gemm_bf16<true, false, true><<<dim3(HID_DIM / 128, N_TOK / 128), 256, 0, stream>>>(
      h2, w1B, b1, nullptr, nullptr, ffn1, N_TOK, HID_DIM, C_DIM, 0);

  // 8) FFN2 + residual(mid) -> fp32 final output.
  gemm_bf16<false, true, false><<<dim3(C_DIM / 128, N_TOK / 128), 256, 0, stream>>>(
      ffn1, w2B, b2, mid, outp, nullptr, N_TOK, C_DIM, HID_DIM, 0);
}
